// ConcreteLayer_12455405158796
// MI455X (gfx1250) — matrix operand order repaired and checked by reading, not compiled, not run on hardware
//
#include <hip/hip_runtime.h>
#include <stdint.h>

// ---------- problem dims (fixed by the reference) ----------
#define B_DIM   4096   // batch (M)
#define IN_DIM  4096   // K
#define OUT_DIM 1024   // N
#define TINY_F  1.17549435e-38f   // np.finfo(float32).tiny

// ---------- vector types (CDNA5 gfx1250, wave32) ----------
typedef __attribute__((ext_vector_type(16))) __bf16        v16bf;
typedef __attribute__((ext_vector_type(8)))  __bf16        v8bf;
typedef __attribute__((ext_vector_type(8)))  float         v8f;
typedef __attribute__((ext_vector_type(4)))  unsigned int  u32x4;
typedef __attribute__((ext_vector_type(8)))  int           i32x8;
typedef __attribute__((ext_vector_type(4)))  int           i32x4;

union FragU { u32x4 q[2]; v16bf v; };   // 32B = one 16-elem bf16 WMMA operand

// =====================================================================
// Kernel 0: x (f32) -> x_bf16, native casts (hardware cvt, RNE).
// =====================================================================
__global__ __launch_bounds__(256) void convert_x_bf16(
    const float* __restrict__ x, __bf16* __restrict__ xh)
{
    const int idx = (blockIdx.x * 256 + threadIdx.x) * 8;
    const float4 f0 = *(const float4*)(x + idx);
    const float4 f1 = *(const float4*)(x + idx + 4);
    v8bf h;
    h[0] = (__bf16)f0.x; h[1] = (__bf16)f0.y;
    h[2] = (__bf16)f0.z; h[3] = (__bf16)f0.w;
    h[4] = (__bf16)f1.x; h[5] = (__bf16)f1.y;
    h[6] = (__bf16)f1.z; h[7] = (__bf16)f1.w;
    *(v8bf*)(xh + idx) = h;
}

// =====================================================================
// Kernel 1: Gumbel transform + row softmax over OUT dim, emit bf16.
// =====================================================================
__global__ __launch_bounds__(256) void concrete_softmax_bf16(
    const float* __restrict__ w, const float* __restrict__ u,
    const float* __restrict__ T, __bf16* __restrict__ s)
{
    __shared__ float red[256];
    const int row  = blockIdx.x;
    const int tid  = threadIdx.x;
    const float invT = 1.0f / T[0];
    const int base = row * OUT_DIM;

    float v[4];
#pragma unroll
    for (int j = 0; j < 4; ++j) {
        const int c = tid + j * 256;
        float uu = u[base + c];
        uu = (1.0f - TINY_F) * uu + TINY_F;
        const float g = -logf(-logf(uu));            // Gumbel(0,1)
        v[j] = (w[base + c] + g) * invT;
    }

    float m = fmaxf(fmaxf(v[0], v[1]), fmaxf(v[2], v[3]));
    red[tid] = m; __syncthreads();
    for (int stp = 128; stp > 0; stp >>= 1) {
        if (tid < stp) red[tid] = fmaxf(red[tid], red[tid + stp]);
        __syncthreads();
    }
    const float rowmax = red[0]; __syncthreads();

    float e[4]; float acc = 0.0f;
#pragma unroll
    for (int j = 0; j < 4; ++j) { e[j] = expf(v[j] - rowmax); acc += e[j]; }
    red[tid] = acc; __syncthreads();
    for (int stp = 128; stp > 0; stp >>= 1) {
        if (tid < stp) red[tid] += red[tid + stp];
        __syncthreads();
    }
    const float rinv = 1.0f / red[0];

#pragma unroll
    for (int j = 0; j < 4; ++j) {
        const int c = tid + j * 256;
        s[base + c] = (__bf16)(e[j] * rinv);
    }
}

// =====================================================================
// TDM descriptor issue: 2D tile load, 2-byte elements, LDS row padding
// row = 64 elems = 32 DWORDs -> pad_interval=4 (32 DW), pad_amount=3 (4 DW)
// => padded LDS row stride = 72 halfwords (bank-skewed, 16B aligned).
// D# layout per CDNA5 ISA ch.8 (groups 2/3 zero => 2D tensor).
// Toolchain here is the 6-arg clang-23 builtin:
//   (uint32x4 g0, int32x8 g1, int32x4 g2, int32x4 g3, int32x8 aux, i32 cpol)
// =====================================================================
#define LST 72

__device__ __forceinline__ void tdm_load_2d_bf16(
    unsigned lds_addr, const void* gptr,
    unsigned tdim0, unsigned tdim1,       // tensor extents (elements)
    unsigned tile0, unsigned tile1,       // tile extents (elements)
    unsigned stride0)                     // row stride (elements)
{
    const unsigned long long ga = (unsigned long long)(uintptr_t)gptr;
    u32x4 g0 = { 1u,                                  // count=1, user mode
                 lds_addr,                            // LDS byte address
                 (unsigned)ga,                        // global_addr[31:0]
                 (unsigned)((ga >> 32) & 0x1FFFFFFu) | (2u << 30) }; // [56:32] | type=2
    i32x8 g1 = { (int)((1u << 16)   |                 // data_size = 2 bytes
                       (1u << 20)   |                 // pad_enable
                       (4u << 22)   |                 // pad_interval: 32 DWORDs
                       (3u << 25)),                   // pad_amount : 4 DWORDs
                 (int)((tdim0 & 0xFFFFu) << 16),      // tensor_dim0[15:0]
                 (int)((tdim0 >> 16) | ((tdim1 & 0xFFFFu) << 16)),
                 (int)((tdim1 >> 16) | (tile0 << 16)),
                 (int)tile1,                          // tile_dim1 | tile_dim2=0
                 (int)stride0,                        // tensor_dim0_stride[31:0]
                 0, 0 };
    i32x4 gz4 = { 0, 0, 0, 0 };
    i32x8 gz8 = { 0, 0, 0, 0, 0, 0, 0, 0 };
    __builtin_amdgcn_tensor_load_to_lds(g0, g1, gz4, gz4, gz8, 0);
}

// =====================================================================
// Kernel 2 (main): out = x_bf16 @ samples_bf16, TDM-fed, double-buffered.
// Block 256 threads (8 wave32). Tile 128(M) x 64(N), K-step 64.
// Waves 4(M) x 2(N): each wave computes 32x32 -> 8 WMMAs per K-step.
// =====================================================================
#define BM 128
#define BN 64
#define BK 64

__global__ __launch_bounds__(256) void wmma_gemm_tdm(
    const __bf16* __restrict__ xh, const __bf16* __restrict__ s,
    float* __restrict__ out)
{
    __shared__ __align__(16) unsigned short Xs[2][BM][LST];
    __shared__ __align__(16) unsigned short Ss[2][BK][LST];

    const int tid  = threadIdx.x;
    const int lane = tid & 31;
    const int wave = tid >> 5;

    const int n_block = blockIdx.x * BN;
    const int m_block = blockIdx.y * BM;

    const int m_base  = (wave >> 1) * 32;        // 0,32,64,96
    const int n_base0 = (wave & 1) * 32;         // 0 or 32
    const int n_base1 = n_base0 + 16;

    // A-frag per-lane addressing (ISA 16-bit A 16x32 layout)
    const int a_m  = lane & 15;
    const int a_k0 = (lane >> 4) * 8;

    const unsigned xs_lds[2] = { (unsigned)(uintptr_t)&Xs[0][0][0],
                                 (unsigned)(uintptr_t)&Xs[1][0][0] };
    const unsigned ss_lds[2] = { (unsigned)(uintptr_t)&Ss[0][0][0],
                                 (unsigned)(uintptr_t)&Ss[1][0][0] };

    v8f acc00 = {}, acc01 = {}, acc10 = {}, acc11 = {};

    const int NT = IN_DIM / BK;   // 64 K-steps

    if (wave == 0) {   // prologue: fetch tile 0 into buffer 0
        tdm_load_2d_bf16(xs_lds[0], xh + m_block * IN_DIM, IN_DIM, B_DIM, BK, BM, IN_DIM);
        tdm_load_2d_bf16(ss_lds[0], s + n_block,           OUT_DIM, IN_DIM, BN, BK, OUT_DIM);
    }

    for (int it = 0; it < NT; ++it) {
        const int cur = it & 1;

        if (wave == 0) {
            if (it + 1 < NT) {   // prefetch next tile into the other buffer
                const int k1 = (it + 1) * BK;
                tdm_load_2d_bf16(xs_lds[cur ^ 1], xh + m_block * IN_DIM + k1,
                                 IN_DIM, B_DIM, BK, BM, IN_DIM);
                tdm_load_2d_bf16(ss_lds[cur ^ 1], s + k1 * OUT_DIM + n_block,
                                 OUT_DIM, IN_DIM, BN, BK, OUT_DIM);
                __builtin_amdgcn_s_wait_tensorcnt(2);  // current tile's 2 loads done
            } else {
                __builtin_amdgcn_s_wait_tensorcnt(0);
            }
        }
        __syncthreads();   // current buffer visible to all waves

        const unsigned short (*X)[LST] = Xs[cur];
        const unsigned short (*S)[LST] = Ss[cur];
#pragma unroll
        for (int ks = 0; ks < 2; ++ks) {
            const int kk = ks * 32;
            FragU fa0, fa1, fb0, fb1;
            fa0.q[0] = *(const u32x4*)&X[m_base + a_m][kk + a_k0];
            fa0.q[1] = *(const u32x4*)&X[m_base + a_m][kk + 16 + a_k0];
            fa1.q[0] = *(const u32x4*)&X[m_base + 16 + a_m][kk + a_k0];
            fa1.q[1] = *(const u32x4*)&X[m_base + 16 + a_m][kk + 16 + a_k0];
#pragma unroll
            for (int j = 0; j < 16; ++j) {
                const int kb = (j < 8) ? (a_k0 + j) : (16 + a_k0 + j - 8);
                fb0.v[j] = __builtin_bit_cast(__bf16, S[kk + kb][n_base0 + a_m]);
                fb1.v[j] = __builtin_bit_cast(__bf16, S[kk + kb][n_base1 + a_m]);
            }

            acc00 = __builtin_amdgcn_wmma_f32_16x16x32_bf16(
                        false, fa0.v, false, fb0.v, (short)0, acc00, false, false);
            acc01 = __builtin_amdgcn_wmma_f32_16x16x32_bf16(
                        false, fa0.v, false, fb1.v, (short)0, acc01, false, false);
            acc10 = __builtin_amdgcn_wmma_f32_16x16x32_bf16(
                        false, fa1.v, false, fb0.v, (short)0, acc10, false, false);
            acc11 = __builtin_amdgcn_wmma_f32_16x16x32_bf16(
                        false, fa1.v, false, fb1.v, (short)0, acc11, false, false);
        }
        __syncthreads();   // reads done before TDM overwrites this buffer
    }

    // epilogue: C/D layout -> VGPR r: M=r (lanes 0-15) / M=r+8 (lanes 16-31)
    const int row0 = m_block + m_base + (lane >> 4) * 8;
    const int col0 = n_block + n_base0 + (lane & 15);
    const int col1 = n_block + n_base1 + (lane & 15);
#pragma unroll
    for (int r = 0; r < 8; ++r) {
        out[(row0 + r) * OUT_DIM + col0]      = acc00[r];
        out[(row0 + r) * OUT_DIM + col1]      = acc01[r];
        out[(row0 + 16 + r) * OUT_DIM + col0] = acc10[r];
        out[(row0 + 16 + r) * OUT_DIM + col1] = acc11[r];
    }
}

// =====================================================================
// Kernel 2 (fallback, small ws): manual staging + on-the-fly f32->bf16.
// Block tile 128x64, K-step 32.
// =====================================================================
#define FBK 32
#define XS_STRIDE 40
#define SS_STRIDE 72

__global__ __launch_bounds__(256) void wmma_gemm_fallback(
    const float* __restrict__ xf, const __bf16* __restrict__ s,
    float* __restrict__ out)
{
    __shared__ __align__(16) unsigned short Xs[BM][XS_STRIDE];
    __shared__ __align__(16) unsigned short Ss[FBK][SS_STRIDE];

    const int tid  = threadIdx.x;
    const int lane = tid & 31;
    const int wave = tid >> 5;

    const int n_block = blockIdx.x * BN;
    const int m_block = blockIdx.y * BM;

    const int m_base  = (wave >> 1) * 32;
    const int n_base0 = (wave & 1) * 32;
    const int n_base1 = n_base0 + 16;

    const int a_m  = lane & 15;
    const int a_k0 = (lane >> 4) * 8;

    const int xr = tid >> 1;
    const int xc = (tid & 1) * 16;
    const int sr = tid >> 3;
    const int sc = (tid & 7) * 8;

    v8f acc00 = {}, acc01 = {}, acc10 = {}, acc11 = {};

    for (int k0 = 0; k0 < IN_DIM; k0 += FBK) {
        const float* xp = xf + (m_block + xr) * IN_DIM + k0 + xc;
        const float4 f0 = *(const float4*)xp;
        const float4 f1 = *(const float4*)(xp + 4);
        const float4 f2 = *(const float4*)(xp + 8);
        const float4 f3 = *(const float4*)(xp + 12);
        v8bf h0, h1;
        h0[0] = (__bf16)f0.x; h0[1] = (__bf16)f0.y;
        h0[2] = (__bf16)f0.z; h0[3] = (__bf16)f0.w;
        h0[4] = (__bf16)f1.x; h0[5] = (__bf16)f1.y;
        h0[6] = (__bf16)f1.z; h0[7] = (__bf16)f1.w;
        h1[0] = (__bf16)f2.x; h1[1] = (__bf16)f2.y;
        h1[2] = (__bf16)f2.z; h1[3] = (__bf16)f2.w;
        h1[4] = (__bf16)f3.x; h1[5] = (__bf16)f3.y;
        h1[6] = (__bf16)f3.z; h1[7] = (__bf16)f3.w;
        *(v8bf*)&Xs[xr][xc]     = h0;
        *(v8bf*)&Xs[xr][xc + 8] = h1;

        const u32x4 ps = *(const u32x4*)(s + (k0 + sr) * OUT_DIM + n_block + sc);
        *(u32x4*)&Ss[sr][sc] = ps;

        __syncthreads();

        FragU fa0, fa1, fb0, fb1;
        fa0.q[0] = *(const u32x4*)&Xs[m_base + a_m][a_k0];
        fa0.q[1] = *(const u32x4*)&Xs[m_base + a_m][16 + a_k0];
        fa1.q[0] = *(const u32x4*)&Xs[m_base + 16 + a_m][a_k0];
        fa1.q[1] = *(const u32x4*)&Xs[m_base + 16 + a_m][16 + a_k0];
#pragma unroll
        for (int j = 0; j < 16; ++j) {
            const int kb = (j < 8) ? (a_k0 + j) : (16 + a_k0 + j - 8);
            fb0.v[j] = __builtin_bit_cast(__bf16, Ss[kb][n_base0 + a_m]);
            fb1.v[j] = __builtin_bit_cast(__bf16, Ss[kb][n_base1 + a_m]);
        }

        acc00 = __builtin_amdgcn_wmma_f32_16x16x32_bf16(
                    false, fa0.v, false, fb0.v, (short)0, acc00, false, false);
        acc01 = __builtin_amdgcn_wmma_f32_16x16x32_bf16(
                    false, fa0.v, false, fb1.v, (short)0, acc01, false, false);
        acc10 = __builtin_amdgcn_wmma_f32_16x16x32_bf16(
                    false, fa1.v, false, fb0.v, (short)0, acc10, false, false);
        acc11 = __builtin_amdgcn_wmma_f32_16x16x32_bf16(
                    false, fa1.v, false, fb1.v, (short)0, acc11, false, false);

        __syncthreads();
    }

    const int row0 = m_block + m_base + (lane >> 4) * 8;
    const int col0 = n_block + n_base0 + (lane & 15);
    const int col1 = n_block + n_base1 + (lane & 15);
#pragma unroll
    for (int r = 0; r < 8; ++r) {
        out[(row0 + r) * OUT_DIM + col0]      = acc00[r];
        out[(row0 + r) * OUT_DIM + col1]      = acc01[r];
        out[(row0 + 16 + r) * OUT_DIM + col0] = acc10[r];
        out[(row0 + 16 + r) * OUT_DIM + col1] = acc11[r];
    }
}

// =====================================================================
extern "C" void kernel_launch(void* const* d_in, const int* in_sizes, int n_in,
                              void* d_out, int out_size, void* d_ws, size_t ws_size,
                              hipStream_t stream) {
    const float* x = (const float*)d_in[0];   // [4096, 4096]
    const float* w = (const float*)d_in[1];   // [4096, 1024]
    const float* u = (const float*)d_in[2];   // [4096, 1024]
    const float* T = (const float*)d_in[3];   // [1]
    float* out = (float*)d_out;               // [4096, 1024]

    const size_t samples_bytes = (size_t)IN_DIM * OUT_DIM * sizeof(__bf16); // 8 MB
    const size_t xbf_bytes     = (size_t)B_DIM * IN_DIM * sizeof(__bf16);   // 32 MB

    __bf16* samples_bf16 = (__bf16*)d_ws;

    concrete_softmax_bf16<<<IN_DIM, 256, 0, stream>>>(w, u, T, samples_bf16);

    dim3 grid(OUT_DIM / BN, B_DIM / BM);      // (16, 32)
    if (ws_size >= samples_bytes + xbf_bytes) {
        __bf16* x_bf16 = (__bf16*)((char*)d_ws + samples_bytes);
        convert_x_bf16<<<(B_DIM * IN_DIM) / (256 * 8), 256, 0, stream>>>(x, x_bf16);
        wmma_gemm_tdm<<<grid, 256, 0, stream>>>(x_bf16, samples_bf16, out);
    } else {
        wmma_gemm_fallback<<<grid, 256, 0, stream>>>(x, samples_bf16, out);
    }
}
